// MultiQueryAttention_44358422233483
// MI455X (gfx1250) — matrix operand order repaired and checked by reading, not compiled, not run on hardware
//
#include <hip/hip_runtime.h>
#include <hip/hip_bf16.h>

typedef __attribute__((ext_vector_type(16))) _Float16 v16h;
typedef __attribute__((ext_vector_type(8)))  _Float16 v8h;
typedef __attribute__((ext_vector_type(8)))  float    v8f;

#define B_   4
#define S_   2048
#define DIM_ 1024
#define H_   16
#define DPH_ 64
#define F_   (H_ * DPH_)   // 1024

// ---------------------------------------------------------------------------
// Workspace layout (bytes)
// ---------------------------------------------------------------------------
#define OFF_XH   ((size_t)0)            // [B*S, DIM]      f16   16 MB
#define OFF_WQT  ((size_t)16777216)     // [H, DPH, DIM]   f16    2 MB
#define OFF_WKT  ((size_t)18874368)     // [DPH, DIM]      f16  128 KB
#define OFF_WVT  ((size_t)19005440)     // [DPH, DIM]      f16  128 KB
#define OFF_WOT  ((size_t)19136512)     // [DIM, F]        f16    2 MB
#define OFF_QB   ((size_t)21233664)     // [B*H, S, DPH]   f16   16 MB (pre-scaled)
#define OFF_KB   ((size_t)38010880)     // [B, S, DPH]     f16    1 MB
#define OFF_VT   ((size_t)39059456)     // [B, DPH, S]     f16    1 MB
#define OFF_OB   ((size_t)40108032)     // [B*S, F]        f16   16 MB

// ---------------------------------------------------------------------------
// Fragment helpers (wave32; A/B in this code's contiguous-K order, C/D per ISA 7.12.2)
// A 16x32 f16:  lane l -> row (l&15); K-run kb..kb+15,
//               kb = (l>>4)*16  (two 16B contiguous loads from row-major K)
// B 32x16 f16:  lane l -> col (l&15); K-run (l>>4)*16 .. +15  (one 32B load)
// C/D 16x16 f32: lane l, vgpr r -> row r + 8*(l>>4), col (l&15)
// ---------------------------------------------------------------------------
__device__ __forceinline__ v16h load_a_frag(const _Float16* p) {
    v8h lo = *(const v8h*)(p);
    v8h hi = *(const v8h*)(p + 8);
    v16h a;
#pragma unroll
    for (int i = 0; i < 8; ++i) { a[i] = lo[i]; a[8 + i] = hi[i]; }
    return a;
}

__device__ __forceinline__ float redmax16(float v) {
#pragma unroll
    for (int m = 1; m < 16; m <<= 1) v = fmaxf(v, __shfl_xor(v, m, 32));
    return v;
}
__device__ __forceinline__ float redsum16(float v) {
#pragma unroll
    for (int m = 1; m < 16; m <<= 1) v += __shfl_xor(v, m, 32);
    return v;
}

// LDS byte-offset of a generic pointer known to live in LDS
__device__ __forceinline__ unsigned lds_off_u32(const void* p) {
    return (unsigned)(unsigned long long)
        (__attribute__((address_space(3))) const char*)p;
}

// Async DMA: 16 bytes per lane, global -> LDS, tracked with ASYNCcnt
__device__ __forceinline__ void async_copy_b128(const void* gsrc, const void* ldst) {
    unsigned           loff = lds_off_u32(ldst);
    unsigned long long ga   = (unsigned long long)gsrc;
    asm volatile("global_load_async_to_lds_b128 %0, %1, off"
                 :: "v"(loff), "v"(ga) : "memory");
}
__device__ __forceinline__ void wait_asynccnt0() {
    asm volatile("s_wait_asynccnt 0x0" ::: "memory");
}

// ---------------------------------------------------------------------------
// Prep: f32 -> f16 convert, and transposing f32 [R][C] -> f16 [C][R]
// ---------------------------------------------------------------------------
__global__ void cvt_f16_kernel(const float* __restrict__ in,
                               _Float16* __restrict__ out, int n) {
    int stride = blockDim.x * gridDim.x;
    for (int i = blockIdx.x * blockDim.x + threadIdx.x; i < n; i += stride)
        out[i] = (_Float16)in[i];
}

__global__ void transpose_f16_kernel(const float* __restrict__ in,
                                     _Float16* __restrict__ out, int R, int C) {
    const float* src = in + (size_t)blockIdx.y * R * C;
    _Float16*    dst = out + (size_t)blockIdx.y * R * C;
    int n = R * C, stride = blockDim.x * gridDim.x;
    for (int i = blockIdx.x * blockDim.x + threadIdx.x; i < n; i += stride) {
        int r = i / C, c = i - r * C;
        dst[(size_t)c * R + r] = (_Float16)src[i];
    }
}

// ---------------------------------------------------------------------------
// QKV projection: per wave one 16x64 strip; K loop manually unrolled x2 with
// two named fragment buffers (no register-rotation copies).
// blockIdx.y: 0..H-1 = Q head (output pre-scaled by 1/sqrt(dph)), H = K, H+1 = V.
// ---------------------------------------------------------------------------
__global__ __launch_bounds__(128) void qkv_gemm_kernel(
    const _Float16* __restrict__ xh,   // [B*S, DIM]
    const _Float16* __restrict__ Wqt,  // [H, DPH, DIM]
    const _Float16* __restrict__ Wkt,  // [DPH, DIM]
    const _Float16* __restrict__ Wvt,  // [DPH, DIM]
    const float* __restrict__ bq, const float* __restrict__ bk,
    const float* __restrict__ bv,
    _Float16* __restrict__ Qb,   // [B*H, S, DPH]
    _Float16* __restrict__ Kb,   // [B, S, DPH]
    _Float16* __restrict__ Vt)   // [B, DPH, S]
{
    const int lane  = threadIdx.x & 31;
    const int wave  = threadIdx.x >> 5;
    const int mtile = blockIdx.x * 4 + wave;     // 0..511
    const int job   = blockIdx.y;                // 0..H+1
    const int row0  = mtile * 16;                // global row in [0, B*S)
    const int b     = row0 / S_;
    const int srow  = row0 - b * S_;

    const _Float16* wt; const float* bias;
    if (job < H_)       { wt = Wqt + (size_t)job * DPH_ * DIM_; bias = bq + job * DPH_; }
    else if (job == H_) { wt = Wkt; bias = bk; }
    else                { wt = Wvt; bias = bv; }

    const int m = lane & 15, half = lane >> 4;
    v8f acc[4];
#pragma unroll
    for (int t = 0; t < 4; ++t) acc[t] = {};

    const _Float16* arow = xh + (size_t)(row0 + m) * DIM_ + half * 16;
    const _Float16* brow = wt + (size_t)m * DIM_ + half * 16;

    // --- software-pipelined K loop, 2 K-steps (64) per iteration ---
    v16h a0, a1, b0[4], b1[4];
    a0 = load_a_frag(arow);
#pragma unroll
    for (int t = 0; t < 4; ++t)
        b0[t] = *(const v16h*)(brow + (size_t)t * 16 * DIM_);

    for (int k0 = 0; k0 < DIM_; k0 += 64) {
        a1 = load_a_frag(arow + k0 + 32);
#pragma unroll
        for (int t = 0; t < 4; ++t)
            b1[t] = *(const v16h*)(brow + (size_t)t * 16 * DIM_ + k0 + 32);
#pragma unroll
        for (int t = 0; t < 4; ++t)
            acc[t] = __builtin_amdgcn_wmma_f32_16x16x32_f16(
                false, a0, false, b0[t], (short)0, acc[t], false, false);
        if (k0 + 64 < DIM_) {
            a0 = load_a_frag(arow + k0 + 64);
#pragma unroll
            for (int t = 0; t < 4; ++t)
                b0[t] = *(const v16h*)(brow + (size_t)t * 16 * DIM_ + k0 + 64);
        }
#pragma unroll
        for (int t = 0; t < 4; ++t)
            acc[t] = __builtin_amdgcn_wmma_f32_16x16x32_f16(
                false, a1, false, b1[t], (short)0, acc[t], false, false);
    }

    const float qscale = (job < H_) ? 0.125f : 1.0f;   // fold 1/sqrt(dph) into Q
#pragma unroll
    for (int t = 0; t < 4; ++t) {
        const int e = t * 16 + m;
        const float bb = bias[e];
#pragma unroll
        for (int r = 0; r < 8; ++r) {
            float val = (acc[t][r] + bb) * qscale;
            int sloc = srow + r + half * 8;
            if (job < H_) {
                Qb[((size_t)(b * H_ + job) * S_ + sloc) * DPH_ + e] = (_Float16)val;
            } else if (job == H_) {
                Kb[((size_t)b * S_ + sloc) * DPH_ + e] = (_Float16)val;
            } else {
                Vt[((size_t)b * DPH_ + e) * S_ + sloc] = (_Float16)val;   // transposed
            }
        }
    }
}

// ---------------------------------------------------------------------------
// Flash attention, block-cooperative:
//  - block (256 thr = 8 waves) owns (b, h, 128 Q rows); each wave one 16-row tile
//  - K tile (32x64) and V tile (64x32) double-buffered in LDS, filled with
//    global_load_async_to_lds_b128 (one b128 per thread per tile),
//    synchronized with s_wait_asynccnt + block barrier
//  - 8 WMMAs per 32-key block (4 scores, 4 PV), online softmax in registers
//  - Q is pre-scaled by 1/sqrt(dph), so scores need no extra multiply
// ---------------------------------------------------------------------------
__global__ __launch_bounds__(256) void mqa_flash_kernel(
    const _Float16* __restrict__ Qb,  // [B*H, S, DPH]  (pre-scaled)
    const _Float16* __restrict__ Kb,  // [B, S, DPH]
    const _Float16* __restrict__ Vt,  // [B, DPH, S]
    _Float16* __restrict__ Ob)        // [B, S, F]
{
    __shared__ __align__(128) _Float16 sK[2][32 * 64];  // [key][feat]
    __shared__ __align__(128) _Float16 sV[2][64 * 32];  // [feat][key]
    __shared__ __align__(128) _Float16 sP[8][16 * 32];  // per-wave P tile

    const int tid  = threadIdx.x;
    const int lane = tid & 31;
    const int wave = tid >> 5;
    const int blk  = blockIdx.x;          // 0 .. B*H*(S/128)-1
    const int grp  = blk & 15;            // q row-group (S/128 == 16)
    const int bh   = blk >> 4;            // b*H + h
    const int b    = bh >> 4;
    const int h    = bh & 15;
    const int st   = grp * 8 + wave;      // 16-row Q tile index

    const int m = lane & 15, half = lane >> 4;

    const _Float16* kbase = Kb + (size_t)b * S_ * DPH_;   // [S][64]
    const _Float16* vbase = Vt + (size_t)b * DPH_ * S_;   // [64][S]

    // per-thread DMA chunk assignment (4KB per tile = 256 x 16B)
    const _Float16* kg = kbase + (size_t)tid * 8;          // + kt*64
    const int vf = tid >> 2, vc = tid & 3;
    const _Float16* vg = vbase + (size_t)vf * S_ + vc * 8; // + kt
    const int kl = tid * 8;                                // half-index in sK[buf]
    const int vl = vf * 32 + vc * 8;                       // half-index in sV[buf]

    // Q fragments (resident for whole kernel)
    const _Float16* qrow = Qb + ((size_t)bh * S_ + st * 16 + m) * DPH_ + half * 16;
    const v16h aq0 = load_a_frag(qrow);
    const v16h aq1 = load_a_frag(qrow + 32);

    v8f o[4];
#pragma unroll
    for (int t = 0; t < 4; ++t) o[t] = {};
    float mrow[8], lrow[8];
#pragma unroll
    for (int r = 0; r < 8; ++r) { mrow[r] = -3.0e38f; lrow[r] = 0.0f; }

    _Float16* pl = &sP[wave][0];

    // prologue: stage tile 0
    int cur = 0;
    async_copy_b128(kg, &sK[0][kl]);
    async_copy_b128(vg, &sV[0][vl]);
    wait_asynccnt0();
    __syncthreads();

    for (int kt = 0; kt < S_; kt += 32) {
        const int nxt = cur ^ 1;
        if (kt + 32 < S_) {                    // stage next tile while computing
            async_copy_b128(kg + (size_t)(kt + 32) * DPH_, &sK[nxt][kl]);
            async_copy_b128(vg + (kt + 32),                &sV[nxt][vl]);
        }

        // ---- scores: two 16x16 fragments over keys kt..kt+31 (from LDS) ----
        v8f s0 = {}, s1 = {};
        {
            const _Float16* kp0 = &sK[cur][m * 64 + half * 16];
            v16h b0 = *(const v16h*)(kp0);
            v16h b1 = *(const v16h*)(kp0 + 32);
            s0 = __builtin_amdgcn_wmma_f32_16x16x32_f16(false, aq0, false, b0, (short)0, s0, false, false);
            s0 = __builtin_amdgcn_wmma_f32_16x16x32_f16(false, aq1, false, b1, (short)0, s0, false, false);
            const _Float16* kp1 = kp0 + 16 * 64;
            v16h b2 = *(const v16h*)(kp1);
            v16h b3 = *(const v16h*)(kp1 + 32);
            s1 = __builtin_amdgcn_wmma_f32_16x16x32_f16(false, aq0, false, b2, (short)0, s1, false, false);
            s1 = __builtin_amdgcn_wmma_f32_16x16x32_f16(false, aq1, false, b3, (short)0, s1, false, false);
        }

        // ---- online softmax (row = r + 8*half in both S and O fragments) ----
        float p0[8], p1[8];
#pragma unroll
        for (int r = 0; r < 8; ++r) {
            float a0 = s0[r], a1 = s1[r];
            float tmax = redmax16(fmaxf(a0, a1));
            float mnew = fmaxf(mrow[r], tmax);
            float alpha = __expf(mrow[r] - mnew);
            float e0 = __expf(a0 - mnew), e1 = __expf(a1 - mnew);
            lrow[r] = lrow[r] * alpha + redsum16(e0 + e1);
            mrow[r] = mnew;
            p0[r] = e0; p1[r] = e1;
#pragma unroll
            for (int t = 0; t < 4; ++t) o[t][r] *= alpha;
        }

        // ---- reshape P (C-layout -> A-layout) via wave-private LDS ----
#pragma unroll
        for (int r = 0; r < 8; ++r) {
            int row = r + half * 8;
            pl[row * 32 + m]      = (_Float16)p0[r];
            pl[row * 32 + 16 + m] = (_Float16)p1[r];
        }
        asm volatile("s_wait_dscnt 0x0" ::: "memory");   // wave-local LDS ordering
        v16h ap = load_a_frag(pl + m * 32 + half * 16);

        // ---- O += P @ V (V from LDS) ----
#pragma unroll
        for (int t = 0; t < 4; ++t) {
            const _Float16* vp = &sV[cur][(t * 16 + m) * 32 + half * 16];
            v16h bf = *(const v16h*)vp;
            o[t] = __builtin_amdgcn_wmma_f32_16x16x32_f16(
                false, ap, false, bf, (short)0, o[t], false, false);
        }

        wait_asynccnt0();      // next tile's DMA complete (this wave's chunks)
        __syncthreads();       // everyone done computing on cur & staging nxt
        cur = nxt;
    }

    // ---- normalize + store concat-head output ----
#pragma unroll
    for (int t = 0; t < 4; ++t) {
#pragma unroll
        for (int r = 0; r < 8; ++r) {
            int s = st * 16 + r + half * 8;
            float val = o[t][r] / lrow[r];
            Ob[((size_t)b * S_ + s) * F_ + h * DPH_ + t * 16 + m] = (_Float16)val;
        }
    }
}

// ---------------------------------------------------------------------------
// Output projection: [B*S, F] x [F, DIM] + bo -> f32 out
// (same 2x-unrolled pipelined K loop, no rotation copies)
// ---------------------------------------------------------------------------
__global__ __launch_bounds__(128) void out_proj_kernel(
    const _Float16* __restrict__ Ob,   // [B*S, F]
    const _Float16* __restrict__ Wot,  // [DIM, F]   (K-contiguous)
    const float* __restrict__ bo,      // [DIM]
    float* __restrict__ out)           // [B*S, DIM]
{
    const int lane  = threadIdx.x & 31;
    const int wave  = threadIdx.x >> 5;
    const int mtile = blockIdx.x * 4 + wave;   // 0..511
    const int n0    = blockIdx.y * 64;
    const int row0  = mtile * 16;
    const int m = lane & 15, half = lane >> 4;

    v8f acc[4];
#pragma unroll
    for (int t = 0; t < 4; ++t) acc[t] = {};

    const _Float16* arow = Ob + (size_t)(row0 + m) * F_ + half * 16;
    const _Float16* brow = Wot + (size_t)(n0 + m) * F_ + half * 16;

    v16h a0, a1, b0[4], b1[4];
    a0 = load_a_frag(arow);
#pragma unroll
    for (int t = 0; t < 4; ++t)
        b0[t] = *(const v16h*)(brow + (size_t)t * 16 * F_);

    for (int k0 = 0; k0 < F_; k0 += 64) {
        a1 = load_a_frag(arow + k0 + 32);
#pragma unroll
        for (int t = 0; t < 4; ++t)
            b1[t] = *(const v16h*)(brow + (size_t)t * 16 * F_ + k0 + 32);
#pragma unroll
        for (int t = 0; t < 4; ++t)
            acc[t] = __builtin_amdgcn_wmma_f32_16x16x32_f16(
                false, a0, false, b0[t], (short)0, acc[t], false, false);
        if (k0 + 64 < F_) {
            a0 = load_a_frag(arow + k0 + 64);
#pragma unroll
            for (int t = 0; t < 4; ++t)
                b0[t] = *(const v16h*)(brow + (size_t)t * 16 * F_ + k0 + 64);
        }
#pragma unroll
        for (int t = 0; t < 4; ++t)
            acc[t] = __builtin_amdgcn_wmma_f32_16x16x32_f16(
                false, a1, false, b1[t], (short)0, acc[t], false, false);
    }

#pragma unroll
    for (int t = 0; t < 4; ++t) {
        const int d = n0 + t * 16 + m;
        const float bb = bo[d];
#pragma unroll
        for (int r = 0; r < 8; ++r) {
            int row = row0 + r + half * 8;
            out[(size_t)row * DIM_ + d] = acc[t][r] + bb;
        }
    }
}

// ---------------------------------------------------------------------------
extern "C" void kernel_launch(void* const* d_in, const int* in_sizes, int n_in,
                              void* d_out, int out_size, void* d_ws, size_t ws_size,
                              hipStream_t stream) {
    const float* x  = (const float*)d_in[0];
    const float* Wq = (const float*)d_in[1];
    const float* bq = (const float*)d_in[2];
    const float* Wk = (const float*)d_in[3];
    const float* bk = (const float*)d_in[4];
    const float* Wv = (const float*)d_in[5];
    const float* bv = (const float*)d_in[6];
    const float* Wo = (const float*)d_in[7];
    const float* bo = (const float*)d_in[8];
    float* out = (float*)d_out;

    char* ws = (char*)d_ws;
    _Float16* xh  = (_Float16*)(ws + OFF_XH);
    _Float16* Wqt = (_Float16*)(ws + OFF_WQT);
    _Float16* Wkt = (_Float16*)(ws + OFF_WKT);
    _Float16* Wvt = (_Float16*)(ws + OFF_WVT);
    _Float16* Wot = (_Float16*)(ws + OFF_WOT);
    _Float16* Qb  = (_Float16*)(ws + OFF_QB);
    _Float16* Kb  = (_Float16*)(ws + OFF_KB);
    _Float16* Vt  = (_Float16*)(ws + OFF_VT);
    _Float16* Ob  = (_Float16*)(ws + OFF_OB);

    // --- prep: convert x, transpose weights into K-contiguous f16 ---
    cvt_f16_kernel<<<2048, 256, 0, stream>>>(x, xh, B_ * S_ * DIM_);
    transpose_f16_kernel<<<dim3(256, H_), 256, 0, stream>>>(Wq, Wqt, DIM_, DPH_);
    transpose_f16_kernel<<<dim3(64, 1),  256, 0, stream>>>(Wk, Wkt, DIM_, DPH_);
    transpose_f16_kernel<<<dim3(64, 1),  256, 0, stream>>>(Wv, Wvt, DIM_, DPH_);
    transpose_f16_kernel<<<dim3(1024, 1), 256, 0, stream>>>(Wo, Wot, F_, DIM_);

    // --- QKV projections (Q per head + shared K/V) ---
    qkv_gemm_kernel<<<dim3(128, H_ + 2), 128, 0, stream>>>(
        xh, Wqt, Wkt, Wvt, bq, bk, bv, Qb, Kb, Vt);

    // --- flash attention: one block per (b, h, 128 Q rows) ---
    mqa_flash_kernel<<<B_ * H_ * (S_ / 128), 256, 0, stream>>>(Qb, Kb, Vt, Ob);

    // --- output projection ---
    out_proj_kernel<<<dim3(128, DIM_ / 64), 128, 0, stream>>>(Ob, Wot, bo, out);
}
